// AngleAwareSelfAttention_52587579572721
// MI455X (gfx1250) — matrix operand order repaired and checked by reading, not compiled, not run on hardware
//
#include <hip/hip_runtime.h>
#include <hip/hip_bf16.h>

typedef __attribute__((ext_vector_type(16))) _Float16 v16h;
typedef __attribute__((ext_vector_type(2)))  _Float16 v2h;
typedef __attribute__((ext_vector_type(8)))  float    v8f;

#define HEADS 8
#define CCH   512
#define NTOK  49
#define NPAD  64
#define QKVW  1536

// LDS strides (elements)
#define XF_STRIDE  513     // f32, odd -> bank friendly
#define XN_STRIDE  520     // f16
#define QKV_STRIDE 1544    // f16

#define QKV_OFF 0                         // bytes; xf (f32) overlays here in phase 1
#define XN_OFF  (64 * QKV_STRIDE * 2)     // = 197632 bytes; attnP overlays here later
#define SMEM_BYTES (XN_OFF + 64 * XN_STRIDE * 2)   // 197632 + 66560 = 264192 B

// ---------------- prep: f16 weight tiles (B-fragment-major) + rel-pos bias ----
__global__ void prep_kernel(const float* __restrict__ wqkv,
                            const float* __restrict__ wout,
                            const float* __restrict__ rel,
                            _Float16* __restrict__ wqT,
                            _Float16* __restrict__ woT,
                            float* __restrict__ biasT) {
    int idx = blockIdx.x * 256 + threadIdx.x;
    const int NQ = CCH * QKVW;        // 786432
    const int NO = CCH * CCH;         // 262144
    const int NB = HEADS * 64 * 64;   // 32768
    if (idx < NQ) {
        int tile = idx >> 9, e = idx & 511;
        int lane = e >> 4, i = e & 15;
        int kt = tile / 96, nt = tile % 96;
        int k = kt * 32 + ((lane >> 4) << 4) + i;
        int n = nt * 16 + (lane & 15);
        wqT[idx] = (_Float16)wqkv[k * QKVW + n];
    } else if (idx < NQ + NO) {
        int j = idx - NQ;
        int tile = j >> 9, e = j & 511;
        int lane = e >> 4, i = e & 15;
        int kt = tile >> 5, nt = tile & 31;
        int k = kt * 32 + ((lane >> 4) << 4) + i;
        int n = nt * 16 + (lane & 15);
        woT[j] = (_Float16)wout[k * CCH + n];
    } else if (idx < NQ + NO + NB) {
        int j = idx - NQ - NO;
        int h = j >> 12, r = j & 4095;
        int i = r >> 6, jj = r & 63;
        float bv = -1e9f;
        if (i < NTOK && jj < NTOK) {
            int hi = i / 7, wi = i % 7, hj = jj / 7, wj = jj % 7;
            bv = rel[((hi - hj + 6) * 13 + (wi - wj + 6)) * HEADS + h];
        }
        biasT[j] = bv;
    }
}

// --------- fragment helpers (contiguous k order, same for A and B tiles) -------
// A (16x32 f16): lane&15 = row; lanes<16 hold K {0..15}, lanes>=16 {16..31}
__device__ __forceinline__ v16h load_a_frag(const _Float16* base, int stride,
                                            int row, int kbase, int lane) {
    int khalf8 = (lane >> 4) << 4;
    v16h a;
#pragma unroll
    for (int v = 0; v < 8; ++v) {
        int k = kbase + khalf8 + (v << 1);
        v2h t = *(const v2h*)(base + row * stride + k);
        a[2 * v] = t[0];
        a[2 * v + 1] = t[1];
    }
    return a;
}

// B fragment from the pre-swizzled tile buffer: 32 contiguous bytes per lane
__device__ __forceinline__ v16h load_b_tile(const _Float16* tiles, int tileId, int lane) {
    union { v16h v; uint4 q[2]; } u;
    const uint4* p = (const uint4*)(tiles + (size_t)tileId * 512 + lane * 16);
    u.q[0] = p[0];
    u.q[1] = p[1];
    return u.v;
}

// --------------------------- fused per-window kernel -------------------------
__global__ __launch_bounds__(256)
void swin_attn_kernel(const float* __restrict__ x, const float* __restrict__ gamma,
                      const float* __restrict__ beta, const _Float16* __restrict__ wqT,
                      const _Float16* __restrict__ woT, const float* __restrict__ biasT,
                      const float* __restrict__ bout, float* __restrict__ out) {
    __shared__ __align__(16) char smem[SMEM_BYTES];
    float*    xf    = (float*)(smem + QKV_OFF);      // [49][513] f32 (phase 1 only)
    _Float16* qkv   = (_Float16*)(smem + QKV_OFF);   // [64][1544] f16 (q|k|v, then out in q)
    _Float16* xn    = (_Float16*)(smem + XN_OFF);    // [64][520] f16 (phase 1-2)
    _Float16* attnP = (_Float16*)(smem + XN_OFF);    // [8][64][64] f16 (phase 3)

    const int tid = threadIdx.x;
    const int lane = tid & 31, wave = tid >> 5;
    const int lane15 = lane & 15, khalf = lane >> 4;
    const int b = blockIdx.x;
    const int xbase = b * (CCH * NTOK);

    // ---- Phase 1a: coalesced (float4) load of x, transposed into LDS [token][chan]
    {
        const float4* x4 = (const float4*)(x + xbase);
        for (int i4 = tid; i4 < (CCH * NTOK) / 4; i4 += 256) {
            float4 vld = x4[i4];
            float vv[4] = {vld.x, vld.y, vld.z, vld.w};
#pragma unroll
            for (int e = 0; e < 4; ++e) {
                int idx = 4 * i4 + e;
                int c = idx / NTOK, p = idx - c * NTOK;
                xf[p * XF_STRIDE + c] = vv[e];
            }
        }
    }
    // zero the M-padding rows of xn (tokens 49..63)
    for (int i = tid; i < (NPAD - NTOK) * CCH; i += 256) {
        int r = NTOK + i / CCH, c = i & (CCH - 1);
        xn[r * XN_STRIDE + c] = (_Float16)0.f;
    }
    __syncthreads();

    // ---- Phase 1b: LayerNorm over channels, one token per wave iteration
    for (int t = wave; t < NTOK; t += 8) {
        float s1 = 0.f, s2 = 0.f;
#pragma unroll
        for (int j = 0; j < 16; ++j) {
            float v = xf[t * XF_STRIDE + lane + 32 * j];
            s1 += v; s2 += v * v;
        }
#pragma unroll
        for (int m = 16; m >= 1; m >>= 1) {
            s1 += __shfl_xor(s1, m, 32);
            s2 += __shfl_xor(s2, m, 32);
        }
        float mu = s1 * (1.f / 512.f);
        float rs = rsqrtf(s2 * (1.f / 512.f) - mu * mu + 1e-5f);
#pragma unroll
        for (int j = 0; j < 16; ++j) {
            int c = lane + 32 * j;
            float v = (xf[t * XF_STRIDE + c] - mu) * rs * gamma[c] + beta[c];
            xn[t * XN_STRIDE + c] = (_Float16)v;
        }
    }
    __syncthreads();

    // ---- Phase 2: QKV GEMM [64,512] x [512,1536] -> qkv LDS (f16)
    // 4x4 register blocking: 4 A-frags x 4 B-frags -> 16 accumulators per K-step
    for (int pass = wave; pass < 24; pass += 8) {
        const int nt0 = pass * 4;
        v8f acc[16];
#pragma unroll
        for (int t = 0; t < 16; ++t) acc[t] = (v8f){};
#pragma unroll 4
        for (int kt = 0; kt < 16; ++kt) {
            v16h Af[4], Bf[4];
#pragma unroll
            for (int mt = 0; mt < 4; ++mt)
                Af[mt] = load_a_frag(xn, XN_STRIDE, mt * 16 + lane15, kt * 32, lane);
#pragma unroll
            for (int j = 0; j < 4; ++j)
                Bf[j] = load_b_tile(wqT, kt * 96 + nt0 + j, lane);
#pragma unroll
            for (int mt = 0; mt < 4; ++mt)
#pragma unroll
                for (int j = 0; j < 4; ++j)
                    acc[mt * 4 + j] = __builtin_amdgcn_wmma_f32_16x16x32_f16(
                        false, Af[mt], false, Bf[j], (short)0, acc[mt * 4 + j], false, false);
        }
#pragma unroll
        for (int mt = 0; mt < 4; ++mt)
#pragma unroll
            for (int j = 0; j < 4; ++j)
#pragma unroll
                for (int v = 0; v < 8; ++v) {
                    int row = mt * 16 + khalf * 8 + v;
                    int col = (nt0 + j) * 16 + lane15;
                    qkv[row * QKV_STRIDE + col] = (_Float16)acc[mt * 4 + j][v];
                }
    }
    __syncthreads();

    // ---- Phase 3: attention, one head per wave
    {
        const int h = wave;
        const float scale = 0.125f;   // 64^-0.5
        // register-cache k^T and v fragments (depend only on nt/kt, shared by all mt)
        v16h kf[8], vf[8];
#pragma unroll
        for (int nt = 0; nt < 4; ++nt)
#pragma unroll
            for (int kt = 0; kt < 2; ++kt) {
                v16h bm;
                int tok = nt * 16 + lane15;
                int cb = 512 + h * 64 + kt * 32 + khalf * 16;
#pragma unroll
                for (int j = 0; j < 8; ++j) {
                    v2h t = *(const v2h*)(qkv + tok * QKV_STRIDE + cb + 2 * j);
                    bm[2 * j] = t[0];
                    bm[2 * j + 1] = t[1];
                }
                kf[nt * 2 + kt] = bm;
            }
#pragma unroll
        for (int nt2 = 0; nt2 < 4; ++nt2)
#pragma unroll
            for (int kt = 0; kt < 2; ++kt) {
                v16h bm;
                int dcol = 1024 + h * 64 + nt2 * 16 + lane15;
#pragma unroll
                for (int j = 0; j < 16; ++j) {
                    int tok = kt * 32 + khalf * 16 + j;
                    bm[j] = qkv[tok * QKV_STRIDE + dcol];
                }
                vf[nt2 * 2 + kt] = bm;
            }

        for (int mt = 0; mt < 4; ++mt) {
            // scores = q @ k^T   (4 N-tiles, K = 64); q A-frags shared across nt
            v16h aq[2];
#pragma unroll
            for (int kt = 0; kt < 2; ++kt)
                aq[kt] = load_a_frag(qkv, QKV_STRIDE, mt * 16 + lane15, h * 64 + kt * 32, lane);
            v8f s[4];
#pragma unroll
            for (int nt = 0; nt < 4; ++nt) {
                v8f acc = {};
#pragma unroll
                for (int kt = 0; kt < 2; ++kt)
                    acc = __builtin_amdgcn_wmma_f32_16x16x32_f16(
                        false, aq[kt], false, kf[nt * 2 + kt], (short)0, acc, false, false);
                s[nt] = acc;
            }
            // softmax (bias already holds -1e9 padding), rows mt*16 + khalf*8 + v
#pragma unroll
            for (int v = 0; v < 8; ++v) {
                int row = mt * 16 + khalf * 8 + v;
                const float* brow = biasT + (h * 64 + row) * 64;
                float vals[4];
#pragma unroll
                for (int nt = 0; nt < 4; ++nt)
                    vals[nt] = s[nt][v] * scale + brow[nt * 16 + lane15];
                float mx = fmaxf(fmaxf(vals[0], vals[1]), fmaxf(vals[2], vals[3]));
#pragma unroll
                for (int m = 8; m >= 1; m >>= 1) mx = fmaxf(mx, __shfl_xor(mx, m, 32));
                float sum = 0.f;
#pragma unroll
                for (int nt = 0; nt < 4; ++nt) { vals[nt] = __expf(vals[nt] - mx); sum += vals[nt]; }
#pragma unroll
                for (int m = 8; m >= 1; m >>= 1) sum += __shfl_xor(sum, m, 32);
                float r = 1.f / sum;
#pragma unroll
                for (int nt = 0; nt < 4; ++nt)
                    attnP[(h * 64 + row) * 64 + nt * 16 + lane15] = (_Float16)(vals[nt] * r);
            }
            // out_rows(mt) = attnP @ v ; overwrite the dead q slots (same head cols)
            v16h ap[2];
#pragma unroll
            for (int kt = 0; kt < 2; ++kt)
                ap[kt] = load_a_frag(attnP + (h * 64) * 64, 64, mt * 16 + lane15, kt * 32, lane);
#pragma unroll
            for (int nt2 = 0; nt2 < 4; ++nt2) {
                v8f acc = {};
#pragma unroll
                for (int kt = 0; kt < 2; ++kt)
                    acc = __builtin_amdgcn_wmma_f32_16x16x32_f16(
                        false, ap[kt], false, vf[nt2 * 2 + kt], (short)0, acc, false, false);
#pragma unroll
                for (int v = 0; v < 8; ++v) {
                    int row = mt * 16 + khalf * 8 + v;
                    qkv[row * QKV_STRIDE + h * 64 + nt2 * 16 + lane15] = (_Float16)acc[v];
                }
            }
        }
    }
    __syncthreads();

    // ---- Phase 4: out-proj [64,512] x [512,512] + b_out + residual -> global
    // one 64x64 register-blocked pass per wave
    {
        const int nt0 = wave * 4;
        v8f acc[16];
#pragma unroll
        for (int t = 0; t < 16; ++t) acc[t] = (v8f){};
#pragma unroll 4
        for (int kt = 0; kt < 16; ++kt) {
            v16h Af[4], Bf[4];
#pragma unroll
            for (int mt = 0; mt < 4; ++mt)
                Af[mt] = load_a_frag(qkv, QKV_STRIDE, mt * 16 + lane15, kt * 32, lane);
#pragma unroll
            for (int j = 0; j < 4; ++j)
                Bf[j] = load_b_tile(woT, kt * 32 + nt0 + j, lane);
#pragma unroll
            for (int mt = 0; mt < 4; ++mt)
#pragma unroll
                for (int j = 0; j < 4; ++j)
                    acc[mt * 4 + j] = __builtin_amdgcn_wmma_f32_16x16x32_f16(
                        false, Af[mt], false, Bf[j], (short)0, acc[mt * 4 + j], false, false);
        }
#pragma unroll
        for (int j = 0; j < 4; ++j) {
            int c = (nt0 + j) * 16 + lane15;
            float bo = bout[c];
#pragma unroll
            for (int mt = 0; mt < 4; ++mt)
#pragma unroll
                for (int v = 0; v < 8; ++v) {
                    int p = mt * 16 + khalf * 8 + v;
                    if (p < NTOK) {
                        int addr = xbase + c * NTOK + p;
                        out[addr] = acc[mt * 4 + j][v] + bo + x[addr];
                    }
                }
        }
    }
}

extern "C" void kernel_launch(void* const* d_in, const int* in_sizes, int n_in,
                              void* d_out, int out_size, void* d_ws, size_t ws_size,
                              hipStream_t stream) {
    const float* x     = (const float*)d_in[0];
    const float* gamma = (const float*)d_in[1];
    const float* beta  = (const float*)d_in[2];
    const float* wqkv  = (const float*)d_in[3];
    const float* rel   = (const float*)d_in[4];
    const float* wout  = (const float*)d_in[5];
    const float* bout  = (const float*)d_in[6];
    float* out = (float*)d_out;

    _Float16* wqT = (_Float16*)d_ws;                        // 512*1536 f16
    _Float16* woT = wqT + CCH * QKVW;                       // 512*512 f16
    float* biasT  = (float*)((char*)d_ws + (size_t)(CCH * QKVW + CCH * CCH) * 2);

    int B = in_sizes[0] / (CCH * NTOK);                     // 2048

    // prep: 786432 + 262144 + 32768 = 1081344 elements = 4224 * 256
    prep_kernel<<<4224, 256, 0, stream>>>(wqkv, wout, rel, wqT, woT, biasT);
    swin_attn_kernel<<<B, 256, 0, stream>>>(x, gamma, beta, wqT, woT, biasT, bout, out);
}
